// GAT_73538430042255
// MI455X (gfx1250) — matrix operand order repaired and checked by reading, not compiled, not run on hardware
//
#include <hip/hip_runtime.h>
#include <hip/hip_bf16.h>

// ---------------------------------------------------------------------------
// GAT (3-layer, inference) for gfx1250 / MI455X.
//   layer0: 128 -> 4 heads x 32, concat, relu
//   layer1: 128 -> 4 heads x 32, concat, relu
//   layer2: 128 -> 1 head  x 128, mean(=identity), none
// Projection GEMMs: v_wmma_f32_16x16x32_f16 (f16 in, f32 acc), with the B
// operand pre-swizzled into WMMA fragment order and staged into LDS via
// global_load_async_to_lds_b128 (ASYNCcnt path).
// Edge softmax: 3 passes (atomic max via uint key, atomic sum of exp,
// weighted atomic scatter), wave32-per-edge on the scatter.
// ---------------------------------------------------------------------------

#define N_NODES 50000
#define N_EDGES 1600000
#define FDIM    128          // IN_DIM == HIDDEN == OUT_DIM == H*D for all layers
#define NEG_SLOPE 0.2f

typedef __attribute__((ext_vector_type(16))) _Float16 v16h;
typedef __attribute__((ext_vector_type(8)))  float    v8f;

// Swizzled B: [nt(8)][k0(4)][lane(32)][j(16)] halves = 16384 halves = 32 KB.
#define BSWZ_HALVES (8 * 4 * 32 * 16)

// ------------------------- helpers -----------------------------------------

// Monotone mapping float -> uint so that unsigned compare == float compare.
__device__ __forceinline__ unsigned fkey(float f) {
    unsigned b = __float_as_uint(f);
    return (b & 0x80000000u) ? ~b : (b | 0x80000000u);
}
__device__ __forceinline__ float funkey(unsigned k) {
    return (k & 0x80000000u) ? __uint_as_float(k & 0x7fffffffu)
                             : __uint_as_float(~k);
}
__device__ __forceinline__ float leaky(float x) {
    return x > 0.0f ? x : NEG_SLOPE * x;
}

// ------------------------- kernels -----------------------------------------

// Convert W (f32, row-major [K=128][N=128]) into f16 WMMA-B fragment order:
//   Bswz[((nt*4 + k0i)*32 + lane)*16 + j] = W[k][col]
//   where col = nt*16 + (lane&15), k = k0i*32 + (lane>=16 ? 8 : 0) + j + (j>=8 ? 8 : 0).
__global__ void pack_B(const float* __restrict__ W, _Float16* __restrict__ Bswz) {
    int u = blockIdx.x * blockDim.x + threadIdx.x;
    if (u >= BSWZ_HALVES) return;
    const int j    = u & 15;
    const int lane = (u >> 4) & 31;
    const int k0i  = (u >> 9) & 3;
    const int nt   = u >> 11;
    const int col  = nt * 16 + (lane & 15);
    const int k    = k0i * 32 + ((lane >= 16) ? 8 : 0) + j + ((j >= 8) ? 8 : 0);
    Bswz[u] = (_Float16)W[k * FDIM + col];
}

// C[M,128] = A[M,128] * B[128,128]; A f32 (converted inline), B pre-swizzled f16.
// Block = 256 threads = 8 wave32; wave w handles N-tile w of one M-tile.
// M (=50000) is an exact multiple of 16, so no tail handling is needed.
__global__ void gat_gemm_wmma(const float* __restrict__ A,
                              const _Float16* __restrict__ Bswz,
                              float* __restrict__ C) {
    __shared__ __attribute__((aligned(32))) _Float16 sB[BSWZ_HALVES]; // 32 KB

    // ---- async-copy the swizzled B operand (32 KB) into LDS --------------
    {
        const int t = threadIdx.x;
        const unsigned lds_base = (unsigned)(size_t)(&sB[0]); // LDS address = low 32 bits
        const unsigned long long g_base = (unsigned long long)(size_t)Bswz;
#pragma unroll
        for (int i = 0; i < 8; ++i) {
            const unsigned off = (unsigned)((i * 256 + t) * 16); // 16B per lane
            const unsigned laddr = lds_base + off;
            const unsigned long long gaddr = g_base + off;
            asm volatile("global_load_async_to_lds_b128 %0, %1, off"
                         :: "v"(laddr), "v"(gaddr) : "memory");
        }
        asm volatile("s_wait_asynccnt 0x0" ::: "memory");
    }
    __syncthreads();

    const int mt   = blockIdx.x;            // M-tile (16 rows)
    const int nt   = threadIdx.x >> 5;      // N-tile (16 cols), 0..7
    const int lane = threadIdx.x & 31;
    const int l15  = lane & 15;
    const bool hi  = lane >= 16;

    const int row = mt * 16 + l15;          // A row held by this lane
    const int col = nt * 16 + l15;          // C column held by this lane
    const float* arow = A + (size_t)row * FDIM;

    v8f acc = {};
#pragma unroll
    for (int k0i = 0; k0i < 4; ++k0i) {
        const int k0 = k0i * 32;
        // A fragment: 16x32 f16. lanes 0-15: elems[0..7]=K+0..7, [8..15]=K+16..23
        //             lanes16-31: elems[0..7]=K+8..15, [8..15]=K+24..31
        v16h a;
        const int ka = k0 + (hi ? 8 : 0);
#pragma unroll
        for (int j = 0; j < 8; ++j) a[j]     = (_Float16)arow[ka + j];
#pragma unroll
        for (int j = 0; j < 8; ++j) a[8 + j] = (_Float16)arow[ka + 16 + j];

        // B fragment: one contiguous 32B read per lane from LDS (ds_load_b128 x2).
        const v16h b = *(const v16h*)&sB[(((nt * 4 + k0i) * 32) + lane) * 16];

        acc = __builtin_amdgcn_wmma_f32_16x16x32_f16(
            /*neg_a=*/false, a, /*neg_b=*/false, b,
            /*c_mod=*/(short)0, acc, /*reuse_a=*/false, /*reuse_b=*/false);
    }

    // D layout: lanes 0-15 -> rows mt*16 + e, lanes 16-31 -> rows mt*16 + 8 + e
    const int rbase = mt * 16 + (hi ? 8 : 0);
#pragma unroll
    for (int e = 0; e < 8; ++e)
        C[(size_t)(rbase + e) * FDIM + col] = acc[e];
}

// el[n,h] = sum_d feat[n,h,d]*al[h,d]; er likewise.
template <int H, int D>
__global__ void gat_elr(const float* __restrict__ feat,
                        const float* __restrict__ al,
                        const float* __restrict__ ar,
                        float* __restrict__ el, float* __restrict__ er) {
    int idx = blockIdx.x * blockDim.x + threadIdx.x;   // over N*H
    if (idx >= N_NODES * H) return;
    const int n = idx / H, h = idx % H;
    const float* f = feat + (size_t)n * (H * D) + h * D;
    float sl = 0.f, sr = 0.f;
#pragma unroll 8
    for (int d = 0; d < D; ++d) {
        float v = f[d];
        sl += v * al[h * D + d];
        sr += v * ar[h * D + d];
    }
    el[idx] = sl;
    er[idx] = sr;
}

// Pass 1: segment max over dst via atomicMax on uint keys.
template <int H>
__global__ void gat_edge_max(const int* __restrict__ src,
                             const int* __restrict__ dst,
                             const float* __restrict__ el,
                             const float* __restrict__ er,
                             unsigned* __restrict__ emaxk) {
    int t = blockIdx.x * blockDim.x + threadIdx.x;     // over E*H
    if (t >= N_EDGES * H) return;
    const int e = t / H, h = t % H;
    const int s = src[e], d = dst[e];
    const float x = leaky(el[s * H + h] + er[d * H + h]);
    atomicMax(&emaxk[d * H + h], fkey(x));
}

// Pass 2: denom[d,h] += exp(e - emax[d,h]).
template <int H>
__global__ void gat_edge_sum(const int* __restrict__ src,
                             const int* __restrict__ dst,
                             const float* __restrict__ el,
                             const float* __restrict__ er,
                             const unsigned* __restrict__ emaxk,
                             float* __restrict__ denom) {
    int t = blockIdx.x * blockDim.x + threadIdx.x;     // over E*H
    if (t >= N_EDGES * H) return;
    const int e = t / H, h = t % H;
    const int s = src[e], d = dst[e];
    const float x = leaky(el[s * H + h] + er[d * H + h]);
    atomicAdd(&denom[d * H + h], __expf(x - funkey(emaxk[d * H + h])));
}

// Pass 3: out[dst] += alpha * feat[src]; one wave32 per edge, 4 floats/lane.
template <int H, int D>
__global__ void gat_edge_scatter(const int* __restrict__ src,
                                 const int* __restrict__ dst,
                                 const float* __restrict__ el,
                                 const float* __restrict__ er,
                                 const unsigned* __restrict__ emaxk,
                                 const float* __restrict__ denom,
                                 const float* __restrict__ feat,
                                 float* __restrict__ out) {
    const int e    = (blockIdx.x * blockDim.x + threadIdx.x) >> 5;
    const int lane = threadIdx.x & 31;
    if (e >= N_EDGES) return;
    const int s = src[e], d = dst[e];

    float alpha[H];
#pragma unroll
    for (int h = 0; h < H; ++h) {
        const float x = leaky(el[s * H + h] + er[d * H + h]);
        alpha[h] = __expf(x - funkey(emaxk[d * H + h])) /
                   (denom[d * H + h] + 1e-9f);
    }
    const float* fs = feat + (size_t)s * FDIM;
    float*       od = out  + (size_t)d * FDIM;
#pragma unroll
    for (int c = 0; c < FDIM / 32; ++c) {
        const int j = lane + 32 * c;       // feature index 0..127
        const int h = j / D;               // constant-folds per c
        atomicAdd(&od[j], fs[j] * alpha[h]);
    }
}

__global__ void relu_inplace(float* __restrict__ x, int n) {
    int i = blockIdx.x * blockDim.x + threadIdx.x;
    if (i < n) x[i] = fmaxf(x[i], 0.0f);
}

// ------------------------- host driver --------------------------------------

extern "C" void kernel_launch(void* const* d_in, const int* in_sizes, int n_in,
                              void* d_out, int out_size, void* d_ws, size_t ws_size,
                              hipStream_t stream) {
    (void)in_sizes; (void)n_in; (void)out_size; (void)ws_size;

    const float* features = (const float*)d_in[0];
    const int*   src      = (const int*)d_in[1];
    const int*   dst      = (const int*)d_in[2];
    const float* W[3]  = {(const float*)d_in[3], (const float*)d_in[6], (const float*)d_in[9]};
    const float* al[3] = {(const float*)d_in[4], (const float*)d_in[7], (const float*)d_in[10]};
    const float* ar[3] = {(const float*)d_in[5], (const float*)d_in[8], (const float*)d_in[11]};
    float* outp = (float*)d_out;

    // Workspace carve (~80 MB).
    const size_t nodeF = (size_t)N_NODES * FDIM;         // 6.4M floats
    const size_t nodeH = (size_t)N_NODES * 4;            // max heads
    char* p = (char*)d_ws;
    float* feat = (float*)p;       p += nodeF * sizeof(float);
    float* h1   = (float*)p;       p += nodeF * sizeof(float);
    float* h2   = (float*)p;       p += nodeF * sizeof(float);
    float* el   = (float*)p;       p += nodeH * sizeof(float);
    float* er   = (float*)p;       p += nodeH * sizeof(float);
    unsigned* emaxk = (unsigned*)p; p += nodeH * sizeof(unsigned);
    float* denom = (float*)p;      p += nodeH * sizeof(float);
    _Float16* Bswz = (_Float16*)p; p += (size_t)BSWZ_HALVES * sizeof(_Float16);

    const float* lin[3]  = {features, h1, h2};
    float*       lout[3] = {h1, h2, outp};
    const int    headsL[3] = {4, 4, 1};

    const int gemmGrid = N_NODES / 16;                   // 3125 (exact)
    const int nElem    = (int)nodeF;

    for (int L = 0; L < 3; ++L) {
        const int H = headsL[L];

        // 1) weights -> f16, swizzled into WMMA B-fragment order
        pack_B<<<(BSWZ_HALVES + 255) / 256, 256, 0, stream>>>(W[L], Bswz);

        // 2) projection GEMM (WMMA, B staged through LDS asynchronously)
        gat_gemm_wmma<<<gemmGrid, 256, 0, stream>>>(lin[L], Bswz, feat);

        // 3) attention logits el/er
        if (H == 4)
            gat_elr<4, 32><<<(N_NODES * 4 + 255) / 256, 256, 0, stream>>>(
                feat, al[L], ar[L], el, er);
        else
            gat_elr<1, 128><<<(N_NODES + 255) / 256, 256, 0, stream>>>(
                feat, al[L], ar[L], el, er);

        // 4) zero accumulators + output
        hipMemsetAsync(emaxk, 0, N_NODES * H * sizeof(unsigned), stream);
        hipMemsetAsync(denom, 0, N_NODES * H * sizeof(float), stream);
        hipMemsetAsync(lout[L], 0, nodeF * sizeof(float), stream);

        // 5) edge softmax passes + scatter
        const int eh = N_EDGES * H;
        if (H == 4) {
            gat_edge_max<4><<<(eh + 255) / 256, 256, 0, stream>>>(src, dst, el, er, emaxk);
            gat_edge_sum<4><<<(eh + 255) / 256, 256, 0, stream>>>(src, dst, el, er, emaxk, denom);
            gat_edge_scatter<4, 32><<<(N_EDGES * 32) / 256, 256, 0, stream>>>(
                src, dst, el, er, emaxk, denom, feat, lout[L]);
        } else {
            gat_edge_max<1><<<(eh + 255) / 256, 256, 0, stream>>>(src, dst, el, er, emaxk);
            gat_edge_sum<1><<<(eh + 255) / 256, 256, 0, stream>>>(src, dst, el, er, emaxk, denom);
            gat_edge_scatter<1, 128><<<(N_EDGES * 32) / 256, 256, 0, stream>>>(
                src, dst, el, er, emaxk, denom, feat, lout[L]);
        }

        // 6) activation (layers 0 and 1 only)
        if (L < 2)
            relu_inplace<<<(nElem + 255) / 256, 256, 0, stream>>>(lout[L], nElem);
    }
}
